// EncoderLayer_44349832298714
// MI455X (gfx1250) — compile-verified, matrix operand order checked by reading, not run on hardware
//
#include <hip/hip_runtime.h>
#include <stdint.h>

// ---------------------------------------------------------------------------
// Problem constants (from reference): B=64, T=512, D=512, I=2D=1024, 4D=2048
// ---------------------------------------------------------------------------
#define BB   64
#define TT   512
#define DD   512
#define II   1024
#define G4D  2048

typedef __bf16 bf16_t;
typedef bf16_t v16bf __attribute__((ext_vector_type(16)));
typedef float  v8f   __attribute__((ext_vector_type(8)));

// fp32 -> bf16 bits, round-to-nearest-even
__device__ __forceinline__ uint16_t f2bf_bits(float f) {
    union { float f; uint32_t u; } c; c.f = f;
    uint32_t u = c.u;
    uint32_t r = (u >> 16) & 1u;
    u += 0x7fffu + r;
    return (uint16_t)(u >> 16);
}

__device__ __forceinline__ float sigm(float x) {
    return 1.0f / (1.0f + __expf(-x));
}

// K-offset pattern of the 16-bit WMMA A/B fragment (16x32 / 32x16):
// lane-half 0 covers K {0..7, 16..23}, half 1 covers K {8..15, 24..31};
// VGPR v holds the contiguous pair starting at koff(v, half).
__device__ __forceinline__ int koff(int v, int half) {
    return (v < 4 ? 2 * v : 16 + 2 * (v - 4)) + 8 * half;
}

// ---------------------------------------------------------------------------
// Pack fp32 weight [K,N] (row-major) into bf16 WMMA-B fragment order:
//   Bp[((nt*(K/32) + kt) * 32 + lane) * 16 + e]
// so each lane's 16 bf16 (32 bytes) of a 32x16 tile are contiguous.
// ---------------------------------------------------------------------------
__global__ void pack_w_kernel(const float* __restrict__ Bsrc,
                              uint16_t* __restrict__ Bp, int K, int N) {
    int gid = blockIdx.x * blockDim.x + threadIdx.x;
    if (gid >= K * N) return;
    int e    = gid & 15;
    int lane = (gid >> 4) & 31;
    int kt   = (gid >> 9) % (K >> 5);
    int nt   = (gid >> 9) / (K >> 5);
    int v = e >> 1, j = e & 1, half = lane >> 4, col = lane & 15;
    int k = kt * 32 + koff(v, half) + j;
    int n = nt * 16 + col;
    Bp[gid] = f2bf_bits(Bsrc[(size_t)k * N + n]);
}

// fp32 -> bf16 elementwise (row-major copy)
__global__ void cvt_bf16_kernel(const float* __restrict__ src,
                                uint16_t* __restrict__ dst, int n) {
    int i = blockIdx.x * blockDim.x + threadIdx.x;
    if (i < n) dst[i] = f2bf_bits(src[i]);
}

__global__ void fill_zero_u32(uint32_t* __restrict__ p, int nwords) {
    int i = blockIdx.x * blockDim.x + threadIdx.x;
    if (i < nwords) p[i] = 0u;
}

// ---------------------------------------------------------------------------
// WMMA GEMM: C[M,N] (fp32) = A_bf16[M,K] (row-major) x Bp (packed) + bias[N]
// One wave computes a 32(M) x 64(N) patch: 2 A-fragments x 4 B-fragments
// -> 8 WMMAs per k-iteration, B traffic amortized over both M-tiles.
// Requires M % 32 == 0 (true here: M = 32768).
// ---------------------------------------------------------------------------
__global__ void gemm_wmma_kernel(const uint16_t* __restrict__ A,
                                 const uint16_t* __restrict__ Bp,
                                 const float* __restrict__ bias,
                                 float* __restrict__ C,
                                 int M, int K, int N) {
    int task = (int)((blockIdx.x * blockDim.x + threadIdx.x) >> 5);
    int lane = threadIdx.x & 31;
    int ngroups = N >> 6;                 // groups of 4 n-tiles
    if (task >= (M >> 5) * ngroups) return;
    int mt2 = task / ngroups;             // 32-row M tile
    int ng  = task % ngroups;

    int colL = lane & 15, half = lane >> 4;
    int row0 = mt2 * 32 + colL;
    int row1 = row0 + 16;
    int KT   = K >> 5;

    v8f acc[2][4];
#pragma unroll
    for (int n = 0; n < 4; ++n) {
        float bv = bias ? bias[ng * 64 + n * 16 + colL] : 0.0f;
#pragma unroll
        for (int v = 0; v < 8; ++v) { acc[0][n][v] = bv; acc[1][n][v] = bv; }
    }

    const uint16_t* Arow0 = A + (size_t)row0 * K;
    const uint16_t* Arow1 = A + (size_t)row1 * K;
    for (int kt = 0; kt < KT; ++kt) {
        if (kt + 1 < KT) {
            __builtin_prefetch((const void*)(Arow0 + (kt + 1) * 32), 0, 3);
            __builtin_prefetch((const void*)(Arow1 + (kt + 1) * 32), 0, 3);
        }
        union { uint32_t u[8]; v16bf v; } Af0, Af1;
#pragma unroll
        for (int v = 0; v < 8; ++v) {
            int k = kt * 32 + koff(v, half);
            Af0.u[v] = *(const uint32_t*)(Arow0 + k);
            Af1.u[v] = *(const uint32_t*)(Arow1 + k);
        }
#pragma unroll
        for (int n = 0; n < 4; ++n) {
            int nt = ng * 4 + n;
            const v16bf Bv =
                *(const v16bf*)(Bp + (((size_t)(nt * KT + kt)) << 9) + (lane << 4));
            acc[0][n] = __builtin_amdgcn_wmma_f32_16x16x32_bf16(
                false, Af0.v, false, Bv, (short)0, acc[0][n], false, false);
            acc[1][n] = __builtin_amdgcn_wmma_f32_16x16x32_bf16(
                false, Af1.v, false, Bv, (short)0, acc[1][n], false, false);
        }
    }

#pragma unroll
    for (int h = 0; h < 2; ++h)
#pragma unroll
        for (int n = 0; n < 4; ++n)
#pragma unroll
            for (int v = 0; v < 8; ++v) {
                int m = mt2 * 32 + h * 16 + v + 8 * half;
                C[(size_t)m * N + ng * 64 + n * 16 + colL] = acc[h][n][v];
            }
}

// ---------------------------------------------------------------------------
// One LSTM time step.  z = xw[:, t, :] + h_prev @ U ; gate order i,f,g,o.
// Each wave owns a 16(batch) x 16(unit) patch and computes all 4 gate tiles
// so the c/h elementwise update fuses in-register.
// grid: 16 blocks x 256 threads = 128 waves = (B/16=4) x (D/16=32) tiles.
// ---------------------------------------------------------------------------
__global__ void lstm_step_kernel(const float* __restrict__ xw,     // [B*T, 4D]
                                 const uint16_t* __restrict__ Up,  // packed [128nt][16kt][512]
                                 float* __restrict__ c,            // [B, D]
                                 const uint16_t* __restrict__ hprev, // [B, D] bf16
                                 uint16_t* __restrict__ hnext,       // [B, D] bf16
                                 uint16_t* __restrict__ seq_bf,      // [B*T, D] or null
                                 float* __restrict__ seq_f32,        // [B*T, D] or null
                                 int t) {
    int wave = (int)((blockIdx.x * blockDim.x + threadIdx.x) >> 5);
    int lane = threadIdx.x & 31;
    int btile = wave >> 5;   // 0..3
    int dtile = wave & 31;   // 0..31
    int colL = lane & 15, half = lane >> 4;

    // init accumulators from precomputed input projection (incl. bias)
    v8f acc[4];
#pragma unroll
    for (int g = 0; g < 4; ++g)
#pragma unroll
        for (int v = 0; v < 8; ++v) {
            int bb = btile * 16 + v + 8 * half;
            int colg = g * DD + dtile * 16 + colL;
            acc[g][v] = xw[((size_t)bb * TT + t) * G4D + colg];
        }

    // z += h_prev @ U   (K = D = 512 -> 16 k-iters of 32)
    const uint16_t* hrow = hprev + (size_t)(btile * 16 + colL) * DD;
    for (int kt = 0; kt < 16; ++kt) {
        union { uint32_t u[8]; v16bf v; } Af;
#pragma unroll
        for (int v = 0; v < 8; ++v)
            Af.u[v] = *(const uint32_t*)(hrow + kt * 32 + koff(v, half));
#pragma unroll
        for (int g = 0; g < 4; ++g) {
            int nt = g * 32 + dtile;          // n-tile inside [4D/16 = 128]
            const v16bf Bv =
                *(const v16bf*)(Up + (((size_t)(nt * 16 + kt)) << 9) + (lane << 4));
            acc[g] = __builtin_amdgcn_wmma_f32_16x16x32_bf16(
                false, Af.v, false, Bv, (short)0, acc[g], false, false);
        }
    }

    // fused gate nonlinearities + state update (relu-LSTM, sigmoid recurrent)
#pragma unroll
    for (int v = 0; v < 8; ++v) {
        int bb = btile * 16 + v + 8 * half;
        int d  = dtile * 16 + colL;
        float iv = acc[0][v], fv = acc[1][v], gv = acc[2][v], ov = acc[3][v];
        float cp = c[(size_t)bb * DD + d];
        float cn = sigm(fv) * cp + sigm(iv) * fmaxf(gv, 0.0f);
        float hn = sigm(ov) * fmaxf(cn, 0.0f);
        c[(size_t)bb * DD + d] = cn;
        uint16_t hbits = f2bf_bits(hn);
        hnext[(size_t)bb * DD + d] = hbits;
        if (seq_bf)  seq_bf[((size_t)bb * TT + t) * DD + d] = hbits;
        if (seq_f32) seq_f32[((size_t)bb * TT + t) * DD + d] = hn;
    }
}

// ---------------------------------------------------------------------------
// LayerNorm over concat(hf,hb) (width 1024) + residual with x.
// One block per (b,t) row, 256 threads x 4 elems.
// ---------------------------------------------------------------------------
__global__ void ln_residual_kernel(const float* __restrict__ x,
                                   const float* __restrict__ hf,
                                   const float* __restrict__ hb,
                                   const float* __restrict__ gamma,
                                   const float* __restrict__ beta,
                                   float* __restrict__ out) {
    int row = blockIdx.x;        // b*T + t
    int tid = threadIdx.x;       // 0..255
    __shared__ float sh[512];

    float vals[4];
    float s = 0.f, q = 0.f;
#pragma unroll
    for (int k = 0; k < 4; ++k) {
        int col = tid + k * 256;
        float v = (col < DD) ? hf[(size_t)row * DD + col]
                             : hb[(size_t)row * DD + (col - DD)];
        vals[k] = v;
        s += v; q += v * v;
    }
    sh[tid] = s; sh[256 + tid] = q;
    __syncthreads();
    for (int off = 128; off > 0; off >>= 1) {
        if (tid < off) {
            sh[tid]       += sh[tid + off];
            sh[256 + tid] += sh[256 + tid + off];
        }
        __syncthreads();
    }
    float mu  = sh[0] * (1.0f / 1024.0f);
    float var = sh[256] * (1.0f / 1024.0f) - mu * mu;
    float rs  = rsqrtf(var + 1e-6f);
#pragma unroll
    for (int k = 0; k < 4; ++k) {
        int col = tid + k * 256;
        out[(size_t)row * II + col] =
            x[(size_t)row * II + col] +
            (vals[k] - mu) * rs * gamma[col] + beta[col];
    }
}

// ---------------------------------------------------------------------------
// Host orchestration
// ---------------------------------------------------------------------------
static inline void* ws_alloc(char* base, size_t& off, size_t bytes) {
    off = (off + 255) & ~(size_t)255;
    void* p = base + off;
    off += bytes;
    return p;
}

extern "C" void kernel_launch(void* const* d_in, const int* in_sizes, int n_in,
                              void* d_out, int out_size, void* d_ws, size_t ws_size,
                              hipStream_t stream) {
    (void)in_sizes; (void)n_in; (void)out_size; (void)ws_size;
    const float* x     = (const float*)d_in[0];
    const float* W1    = (const float*)d_in[1];
    const float* U1    = (const float*)d_in[2];
    const float* b1    = (const float*)d_in[3];
    const float* Wf    = (const float*)d_in[4];
    const float* Uf    = (const float*)d_in[5];
    const float* bf    = (const float*)d_in[6];
    const float* Wb    = (const float*)d_in[7];
    const float* Ub    = (const float*)d_in[8];
    const float* bb    = (const float*)d_in[9];
    const float* gamma = (const float*)d_in[10];
    const float* beta  = (const float*)d_in[11];
    float* out = (float*)d_out;

    char* base = (char*)d_ws;
    size_t off = 0;
    const size_t MBT = (size_t)BB * TT;          // 32768 rows

    uint16_t* W1p   = (uint16_t*)ws_alloc(base, off, (size_t)II * G4D * 2);
    uint16_t* U1p   = (uint16_t*)ws_alloc(base, off, (size_t)DD * G4D * 2);
    uint16_t* Wfp   = (uint16_t*)ws_alloc(base, off, (size_t)DD * G4D * 2);
    uint16_t* Ufp   = (uint16_t*)ws_alloc(base, off, (size_t)DD * G4D * 2);
    uint16_t* Wbp   = (uint16_t*)ws_alloc(base, off, (size_t)DD * G4D * 2);
    uint16_t* Ubp   = (uint16_t*)ws_alloc(base, off, (size_t)DD * G4D * 2);
    uint16_t* x_bf  = (uint16_t*)ws_alloc(base, off, MBT * II * 2);
    uint16_t* h1_bf = (uint16_t*)ws_alloc(base, off, MBT * DD * 2);
    float*    hf32  = (float*)   ws_alloc(base, off, MBT * DD * 4);
    float*    hb32  = (float*)   ws_alloc(base, off, MBT * DD * 4);
    float*    xwA   = (float*)   ws_alloc(base, off, MBT * G4D * 4); // xw1 then xwf
    float*    xwB   = (float*)   ws_alloc(base, off, MBT * G4D * 4); // xwb
    float*    c1    = (float*)   ws_alloc(base, off, (size_t)BB * DD * 4);
    float*    cf    = (float*)   ws_alloc(base, off, (size_t)BB * DD * 4);
    float*    cb    = (float*)   ws_alloc(base, off, (size_t)BB * DD * 4);
    uint16_t* hA[2] = { (uint16_t*)ws_alloc(base, off, (size_t)BB * DD * 2),
                        (uint16_t*)ws_alloc(base, off, (size_t)BB * DD * 2) };
    uint16_t* hF[2] = { (uint16_t*)ws_alloc(base, off, (size_t)BB * DD * 2),
                        (uint16_t*)ws_alloc(base, off, (size_t)BB * DD * 2) };
    uint16_t* hB[2] = { (uint16_t*)ws_alloc(base, off, (size_t)BB * DD * 2),
                        (uint16_t*)ws_alloc(base, off, (size_t)BB * DD * 2) };

    // ---- zero initial states -------------------------------------------------
    {
        int cw = BB * DD;                 // fp32 words per c buffer
        int hw = (BB * DD) / 2;           // u32 words per bf16 h buffer
        fill_zero_u32<<<(cw + 255) / 256, 256, 0, stream>>>((uint32_t*)c1, cw);
        fill_zero_u32<<<(cw + 255) / 256, 256, 0, stream>>>((uint32_t*)cf, cw);
        fill_zero_u32<<<(cw + 255) / 256, 256, 0, stream>>>((uint32_t*)cb, cw);
        fill_zero_u32<<<(hw + 255) / 256, 256, 0, stream>>>((uint32_t*)hA[0], hw);
        fill_zero_u32<<<(hw + 255) / 256, 256, 0, stream>>>((uint32_t*)hF[0], hw);
        fill_zero_u32<<<(hw + 255) / 256, 256, 0, stream>>>((uint32_t*)hB[0], hw);
    }

    // ---- pack weights into WMMA-B fragment order ----------------------------
    pack_w_kernel<<<(II * G4D + 255) / 256, 256, 0, stream>>>(W1, W1p, II, G4D);
    pack_w_kernel<<<(DD * G4D + 255) / 256, 256, 0, stream>>>(U1, U1p, DD, G4D);
    pack_w_kernel<<<(DD * G4D + 255) / 256, 256, 0, stream>>>(Wf, Wfp, DD, G4D);
    pack_w_kernel<<<(DD * G4D + 255) / 256, 256, 0, stream>>>(Uf, Ufp, DD, G4D);
    pack_w_kernel<<<(DD * G4D + 255) / 256, 256, 0, stream>>>(Wb, Wbp, DD, G4D);
    pack_w_kernel<<<(DD * G4D + 255) / 256, 256, 0, stream>>>(Ub, Ubp, DD, G4D);

    // ---- x -> bf16 -----------------------------------------------------------
    {
        int n = (int)(MBT * II);
        cvt_bf16_kernel<<<(n + 255) / 256, 256, 0, stream>>>(x, x_bf, n);
    }

    // ---- layer 1: xw1 = x @ W1 + b1 -----------------------------------------
    {
        int tasks = (int)(MBT / 32) * (G4D / 64);
        gemm_wmma_kernel<<<(tasks + 7) / 8, 256, 0, stream>>>(
            x_bf, W1p, b1, xwA, (int)MBT, II, G4D);
    }
    // sequential recurrence, layer 1
    for (int t = 0; t < TT; ++t)
        lstm_step_kernel<<<16, 256, 0, stream>>>(
            xwA, U1p, c1, hA[t & 1], hA[(t + 1) & 1], h1_bf, nullptr, t);

    // ---- layer 2 projections (reuse xwA for xwf) ----------------------------
    {
        int tasks = (int)(MBT / 32) * (G4D / 64);
        gemm_wmma_kernel<<<(tasks + 7) / 8, 256, 0, stream>>>(
            h1_bf, Wfp, bf, xwA, (int)MBT, DD, G4D);
        gemm_wmma_kernel<<<(tasks + 7) / 8, 256, 0, stream>>>(
            h1_bf, Wbp, bb, xwB, (int)MBT, DD, G4D);
    }
    // bidirectional recurrence (fwd t ascending, bwd t descending)
    for (int s = 0; s < TT; ++s) {
        lstm_step_kernel<<<16, 256, 0, stream>>>(
            xwA, Ufp, cf, hF[s & 1], hF[(s + 1) & 1], nullptr, hf32, s);
        lstm_step_kernel<<<16, 256, 0, stream>>>(
            xwB, Ubp, cb, hB[s & 1], hB[(s + 1) & 1], nullptr, hb32, TT - 1 - s);
    }

    // ---- LayerNorm(concat) + residual ---------------------------------------
    ln_residual_kernel<<<(int)MBT, 256, 0, stream>>>(x, hf32, hb32, gamma, beta, out);
}
